// _SelfAttentionHead_84877143703999
// MI455X (gfx1250) — matrix operand order repaired and checked by reading, not compiled, not run on hardware
//
#include <hip/hip_runtime.h>

typedef __bf16 bf16_t;
typedef __attribute__((ext_vector_type(16))) __bf16 v16bf;
typedef __attribute__((ext_vector_type(8)))  __bf16 v8bf;
typedef __attribute__((ext_vector_type(4)))  __bf16 v4bf;
typedef __attribute__((ext_vector_type(8)))  float  v8f;

#define B_   4
#define S_   2048
#define H_   2048
#define NH_  16
#define HD_  128
#define SCALE_ 0.08838834764831845f   // 1/sqrt(128)
#define NEG_BIG (-3.0e38f)

__device__ __forceinline__ v8f zero8() {
  v8f v = {0.f,0.f,0.f,0.f,0.f,0.f,0.f,0.f};
  return v;
}
__device__ __forceinline__ v16bf cat16(v8bf lo, v8bf hi) {
  return __builtin_shufflevector(lo, hi, 0,1,2,3,4,5,6,7,8,9,10,11,12,13,14,15);
}

// CDNA5 async global->LDS copy (16B per lane), tracked by ASYNCcnt.
// lds generic pointer: low 32 bits are the LDS byte offset (ISA flat->LDS rule).
__device__ __forceinline__ void async_load_b128(const void* g, void* lds) {
  asm volatile("global_load_async_to_lds_b128 %0, %1, off"
               :: "v"((unsigned)(uintptr_t)lds), "v"(g) : "memory");
}
__device__ __forceinline__ void wait_async0() {
  asm volatile("s_wait_asynccnt 0x0" ::: "memory");
}

// ---------------------------------------------------------------------------
// fp32 -> bf16 conversion (vectorized x4)
// ---------------------------------------------------------------------------
__global__ __launch_bounds__(256)
void cvt_f32_to_bf16(const float4* __restrict__ in, v4bf* __restrict__ out, int n4) {
  int i = blockIdx.x * 256 + threadIdx.x;
  if (i >= n4) return;
  float4 v = in[i];
  v4bf o;
  o[0] = (bf16_t)v.x; o[1] = (bf16_t)v.y; o[2] = (bf16_t)v.z; o[3] = (bf16_t)v.w;
  out[i] = o;
}

// ---------------------------------------------------------------------------
// bf16 GEMM: C(MxN) = A(MxK, row-major) * Bw(NxK, row-major)^T
// M=8192, N=K=2048. Block tile 128x128, K-step 32, double-buffered LDS filled
// by GLOBAL_LOAD_ASYNC_TO_LDS_B128 (no VGPR staging).
// 256 threads = 8 waves in 2x4 layout; wave tile 64x32 (4x2 WMMA accums).
// headsplit: map (m,n) -> (b, h, s, d) layout ((b*NH+h)*S+s)*HD+d
// ---------------------------------------------------------------------------
#define GPAD 40   // halves per LDS row (conflict-free for 16B lane windows)

__global__ __launch_bounds__(256)
void gemm_bf16_wmma(const bf16_t* __restrict__ A, const bf16_t* __restrict__ Bw,
                    float* __restrict__ Cf, bf16_t* __restrict__ Cb, int headsplit)
{
  __shared__ bf16_t As[2][128 * GPAD];
  __shared__ bf16_t Bs[2][128 * GPAD];

  const int K = 2048, N = 2048;
  const int m0 = blockIdx.x * 128;
  const int n0 = blockIdx.y * 128;
  const int lane = threadIdx.x;
  const int w    = threadIdx.y;
  const int tid  = w * 32 + lane;
  const int l15  = lane & 15;
  const int lh   = lane >> 4;
  const int wm   = (w >> 2) * 64;   // wave row offset in tile
  const int wn   = (w & 3)  * 32;   // wave col offset in tile

  v8f acc[4][2];
#pragma unroll
  for (int mt = 0; mt < 4; ++mt) { acc[mt][0] = zero8(); acc[mt][1] = zero8(); }

  // Each thread stages 2 A-chunks + 2 B-chunks (16B each) per K-step.
  const int r0 = tid >> 2,         kc0 = (tid & 3) * 8;
  const int r1 = (tid + 256) >> 2, kc1 = ((tid + 256) & 3) * 8;

  const bf16_t* Abase = A  + (long)m0 * K;
  const bf16_t* Bbase = Bw + (long)n0 * K;

  // prologue: stage K-step 0 into buffer 0 via async copies
  async_load_b128(Abase + (long)r0 * K + kc0, &As[0][r0 * GPAD + kc0]);
  async_load_b128(Abase + (long)r1 * K + kc1, &As[0][r1 * GPAD + kc1]);
  async_load_b128(Bbase + (long)r0 * K + kc0, &Bs[0][r0 * GPAD + kc0]);
  async_load_b128(Bbase + (long)r1 * K + kc1, &Bs[0][r1 * GPAD + kc1]);
  wait_async0();
  __syncthreads();

  const int NK = K / 32;
  for (int kt = 0; kt < NK; ++kt) {
    const int buf = kt & 1;
    if (kt + 1 < NK) {  // async-stage next K-step while computing this one
      const int ko = (kt + 1) * 32;
      const int nb = buf ^ 1;
      async_load_b128(Abase + (long)r0 * K + ko + kc0, &As[nb][r0 * GPAD + kc0]);
      async_load_b128(Abase + (long)r1 * K + ko + kc1, &As[nb][r1 * GPAD + kc1]);
      async_load_b128(Bbase + (long)r0 * K + ko + kc0, &Bs[nb][r0 * GPAD + kc0]);
      async_load_b128(Bbase + (long)r1 * K + ko + kc1, &Bs[nb][r1 * GPAD + kc1]);
    }
    // preload ALL fragments, then stream 8 back-to-back WMMAs
    v16bf bfr[2], afr[4];
#pragma unroll
    for (int nt = 0; nt < 2; ++nt) {
      const bf16_t* p = &Bs[buf][(wn + nt * 16 + l15) * GPAD];
      bfr[nt] = cat16(*(const v8bf*)(p + lh * 8), *(const v8bf*)(p + 16 + lh * 8));
    }
#pragma unroll
    for (int mt = 0; mt < 4; ++mt) {
      const bf16_t* p = &As[buf][(wm + mt * 16 + l15) * GPAD];
      afr[mt] = cat16(*(const v8bf*)(p + lh * 8), *(const v8bf*)(p + 16 + lh * 8));
    }
#pragma unroll
    for (int mt = 0; mt < 4; ++mt) {
#pragma unroll
      for (int nt = 0; nt < 2; ++nt) {
        acc[mt][nt] = __builtin_amdgcn_wmma_f32_16x16x32_bf16(
            false, afr[mt], false, bfr[nt], (short)0, acc[mt][nt], false, false);
      }
    }
    if (kt + 1 < NK) {
      wait_async0();      // our async stores into buf^1 have landed in LDS
      __syncthreads();    // everyone's have
    }
  }

  // Epilogue: C/D layout: VGPR r -> row r (lanes 0-15) / r+8 (lanes 16-31), col = lane&15
#pragma unroll
  for (int mt = 0; mt < 4; ++mt) {
#pragma unroll
    for (int nt = 0; nt < 2; ++nt) {
      const int gn = n0 + wn + nt * 16 + l15;
#pragma unroll
      for (int r = 0; r < 8; ++r) {
        const int gm = m0 + wm + mt * 16 + r + 8 * lh;
        float v = acc[mt][nt][r];
        int idx;
        if (headsplit) {
          const int b = gm >> 11, s = gm & 2047, h = gn >> 7, d = gn & 127;
          idx = ((b * NH_ + h) * S_ + s) * HD_ + d;
        } else {
          idx = gm * N + gn;
        }
        if (Cf) Cf[idx] = v;
        if (Cb) Cb[idx] = (bf16_t)v;
      }
    }
  }
}

// ---------------------------------------------------------------------------
// In-place RoPE on bf16 (b,h,s,d) tensor; optionally also writes fp32 copy.
// One thread handles the (d, d+64) pair -> no in-place hazard.
// ---------------------------------------------------------------------------
__global__ __launch_bounds__(256)
void rope_inplace(bf16_t* __restrict__ qk, float* __restrict__ kout,
                  const float* __restrict__ cosp, const float* __restrict__ sinp)
{
  const int i  = blockIdx.x * 256 + threadIdx.x;   // B*NH*S*64 threads
  const int d  = i & 63;
  const int s  = (i >> 6) & (S_ - 1);
  const int bh = i >> 17;
  const int off = (bh * S_ + s) * HD_ + d;
  const float x1 = (float)qk[off];
  const float x2 = (float)qk[off + 64];
  const float c1 = cosp[s * HD_ + d],      s1 = sinp[s * HD_ + d];
  const float c2 = cosp[s * HD_ + d + 64], s2 = sinp[s * HD_ + d + 64];
  const float o1 = x1 * c1 - x2 * s1;   // rotate_half: first half gets -x[d+64]
  const float o2 = x2 * c2 + x1 * s2;
  qk[off]      = (bf16_t)o1;
  qk[off + 64] = (bf16_t)o2;
  if (kout) { kout[off] = o1; kout[off + 64] = o2; }
}

// ---------------------------------------------------------------------------
// Causal flash attention, bf16 WMMA, fp32 online softmax.
// Block = (32,8): 128 query rows per block (wave w owns rows w*16..w*16+15),
// key tiles of 64. Q fragments loaded directly from global in A-layout.
// K tile staged via async global->LDS; V tile transposed through VGPRs.
// O written bf16 in (b, s, h*HD+d) row-major for the output GEMM.
// ---------------------------------------------------------------------------
#define KPAD 136   // Ks row stride (d-dim 128 + 8)
#define VPAD 72    // Vt/Ps row stride (key-dim 64 + 8)

__global__ __launch_bounds__(256)
void attn_fwd_wmma(const bf16_t* __restrict__ Q, const bf16_t* __restrict__ Kk,
                   const bf16_t* __restrict__ V, bf16_t* __restrict__ O)
{
  __shared__ bf16_t Ks[64 * KPAD];    // [key][d]
  __shared__ bf16_t Vt[128 * VPAD];   // [d][key]  (transposed at load)
  __shared__ bf16_t Ps[128 * VPAD];   // [qrow][key]

  const int bh  = blockIdx.x;            // b*NH + h
  const int qt  = blockIdx.y;            // 128-row query tile
  const int b   = bh >> 4, h = bh & 15;
  const int lane = threadIdx.x, w = threadIdx.y;
  const int tid  = w * 32 + lane;
  const int l15  = lane & 15, lh = lane >> 4;

  // Preload Q fragments (A-matrix layout) straight from global memory.
  const bf16_t* Qrow = Q + ((long)bh * S_ + qt * 128 + w * 16 + l15) * HD_;
  v16bf qa[4];
#pragma unroll
  for (int ks = 0; ks < 4; ++ks)
    qa[ks] = cat16(*(const v8bf*)(Qrow + ks * 32 + lh * 8),
                   *(const v8bf*)(Qrow + ks * 32 + 16 + lh * 8));

  float rmax[8], rsum[8];
  v8f aco[8];
#pragma unroll
  for (int r = 0; r < 8; ++r) { rmax[r] = NEG_BIG; rsum[r] = 0.f; }
#pragma unroll
  for (int nt = 0; nt < 8; ++nt) aco[nt] = zero8();

  const int nkt = 2 * qt + 2;   // 64-key tiles covering keys 0..(qt+1)*128-1
  for (int kt = 0; kt < nkt; ++kt) {
    // --- stage K tile via async copies; V tile transposed via VGPRs ---
    const bf16_t* Kb = Kk + ((long)bh * S_ + kt * 64) * HD_;
    const bf16_t* Vb = V  + ((long)bh * S_ + kt * 64) * HD_;
#pragma unroll
    for (int i = 0; i < 4; ++i) {
      const int c = tid + i * 256;
      const int row = c >> 4, dc = (c & 15) * 8;
      async_load_b128(Kb + row * HD_ + dc, &Ks[row * KPAD + dc]);
    }
#pragma unroll
    for (int i = 0; i < 4; ++i) {
      const int c = tid + i * 256;
      const int row = c >> 4, dc = (c & 15) * 8;
      v8bf vv = *(const v8bf*)(Vb + row * HD_ + dc);
#pragma unroll
      for (int j = 0; j < 8; ++j) Vt[(dc + j) * VPAD + row] = vv[j];
    }
    wait_async0();
    __syncthreads();

    // --- scores S = Q K^T  (4 d-steps x 4 key n-tiles) ---
    v8f acs[4];
#pragma unroll
    for (int nt = 0; nt < 4; ++nt) acs[nt] = zero8();
#pragma unroll
    for (int ks = 0; ks < 4; ++ks) {
      v16bf bk[4];
#pragma unroll
      for (int nt = 0; nt < 4; ++nt) {
        const bf16_t* kp = &Ks[(nt * 16 + l15) * KPAD + ks * 32];
        bk[nt] = cat16(*(const v8bf*)(kp + lh * 8), *(const v8bf*)(kp + 16 + lh * 8));
      }
#pragma unroll
      for (int nt = 0; nt < 4; ++nt) {
        acs[nt] = __builtin_amdgcn_wmma_f32_16x16x32_bf16(
            false, qa[ks], false, bk[nt], (short)0, acs[nt], false, false);
      }
    }

    // --- scale + causal mask (only tiles straddling the diagonal) ---
    const bool diag = (kt >= 2 * qt);
    const int  coff = (kt - 2 * qt) * 64;   // key offset relative to row block
#pragma unroll
    for (int nt = 0; nt < 4; ++nt) {
#pragma unroll
      for (int r = 0; r < 8; ++r) {
        float sc = acs[nt][r] * SCALE_;
        if (diag && (coff + nt * 16 + l15) > (w * 16 + r + 8 * lh)) sc = NEG_BIG;
        acs[nt][r] = sc;
      }
    }

    // --- online softmax (per-row stats across 16-lane half) ---
    float fac[8], nmax[8];
#pragma unroll
    for (int r = 0; r < 8; ++r) {
      float m = acs[0][r];
#pragma unroll
      for (int nt = 1; nt < 4; ++nt) m = fmaxf(m, acs[nt][r]);
#pragma unroll
      for (int off = 8; off >= 1; off >>= 1) m = fmaxf(m, __shfl_xor(m, off, 32));
      m = fmaxf(m, rmax[r]);
      fac[r]  = __expf(rmax[r] - m);   // 0 on first tile (exp(-big))
      nmax[r] = m;
    }
#pragma unroll
    for (int nt = 0; nt < 4; ++nt) {
#pragma unroll
      for (int r = 0; r < 8; ++r) acs[nt][r] = __expf(acs[nt][r] - nmax[r]);
    }
#pragma unroll
    for (int r = 0; r < 8; ++r) {
      float ssum = 0.f;
#pragma unroll
      for (int nt = 0; nt < 4; ++nt) ssum += acs[nt][r];
#pragma unroll
      for (int off = 8; off >= 1; off >>= 1) ssum += __shfl_xor(ssum, off, 32);
      rsum[r] = rsum[r] * fac[r] + ssum;
      rmax[r] = nmax[r];
    }
#pragma unroll
    for (int nt = 0; nt < 8; ++nt) {
#pragma unroll
      for (int r = 0; r < 8; ++r) aco[nt][r] *= fac[r];
    }

    // --- write P (bf16) for this wave's own rows; read back as A fragments.
    //     Wave-private rows -> only within-wave LDS ordering needed. ---
#pragma unroll
    for (int nt = 0; nt < 4; ++nt) {
#pragma unroll
      for (int r = 0; r < 8; ++r)
        Ps[(w * 16 + r + 8 * lh) * VPAD + nt * 16 + l15] = (bf16_t)acs[nt][r];
    }

    // --- O += P V  (2 key-steps x 8 d n-tiles) ---
#pragma unroll
    for (int ks = 0; ks < 2; ++ks) {
      const bf16_t* pp = &Ps[(w * 16 + l15) * VPAD + ks * 32];
      v16bf pa = cat16(*(const v8bf*)(pp + lh * 8), *(const v8bf*)(pp + 16 + lh * 8));
      v16bf bv[8];
#pragma unroll
      for (int nt = 0; nt < 8; ++nt) {
        const bf16_t* vp = &Vt[(nt * 16 + l15) * VPAD + ks * 32];
        bv[nt] = cat16(*(const v8bf*)(vp + lh * 8), *(const v8bf*)(vp + 16 + lh * 8));
      }
#pragma unroll
      for (int nt = 0; nt < 8; ++nt) {
        aco[nt] = __builtin_amdgcn_wmma_f32_16x16x32_bf16(
            false, pa, false, bv[nt], (short)0, aco[nt], false, false);
      }
    }
    __syncthreads();   // before next iteration overwrites Ks/Vt
  }

  // --- epilogue: normalize and store (b, s, h*HD + d) bf16 ---
#pragma unroll
  for (int nt = 0; nt < 8; ++nt) {
#pragma unroll
    for (int r = 0; r < 8; ++r) {
      const int s   = qt * 128 + w * 16 + r + 8 * lh;
      const int col = h * HD_ + nt * 16 + l15;
      const float o = aco[nt][r] / rsum[r];
      O[((long)b * S_ + s) * H_ + col] = (bf16_t)o;
    }
  }
}

// ---------------------------------------------------------------------------
// Host-side launch sequence
// ---------------------------------------------------------------------------
extern "C" void kernel_launch(void* const* d_in, const int* in_sizes, int n_in,
                              void* d_out, int out_size, void* d_ws, size_t ws_size,
                              hipStream_t stream)
{
  (void)in_sizes; (void)n_in; (void)out_size; (void)ws_size;
  const float* x    = (const float*)d_in[0];
  const float* cosp = (const float*)d_in[1];
  const float* sinp = (const float*)d_in[2];
  // d_in[3] = mask (causal, handled analytically)
  const float* Wq   = (const float*)d_in[4];
  const float* Wk   = (const float*)d_in[5];
  const float* Wv   = (const float*)d_in[6];
  const float* Wo   = (const float*)d_in[7];

  const size_t NX = (size_t)B_ * S_ * H_;   // 16,777,216
  const size_t NW = (size_t)H_ * H_;        //  4,194,304

  float* y_out = (float*)d_out;
  float* k_out = y_out + NX;
  float* v_out = k_out + NX;

  bf16_t* xb  = (bf16_t*)d_ws;
  bf16_t* wqb = xb  + NX;
  bf16_t* wkb = wqb + NW;
  bf16_t* wvb = wkb + NW;
  bf16_t* wob = wvb + NW;
  bf16_t* qb  = wob + NW;
  bf16_t* kb  = qb  + NX;
  bf16_t* vb  = kb  + NX;
  bf16_t* ob  = vb  + NX;   // total ws use: ~192 MiB

  // 1) fp32 -> bf16 staging
  cvt_f32_to_bf16<<<(int)(NX / 4 / 256), 256, 0, stream>>>((const float4*)x,  (v4bf*)xb,  (int)(NX / 4));
  cvt_f32_to_bf16<<<(int)(NW / 4 / 256), 256, 0, stream>>>((const float4*)Wq, (v4bf*)wqb, (int)(NW / 4));
  cvt_f32_to_bf16<<<(int)(NW / 4 / 256), 256, 0, stream>>>((const float4*)Wk, (v4bf*)wkb, (int)(NW / 4));
  cvt_f32_to_bf16<<<(int)(NW / 4 / 256), 256, 0, stream>>>((const float4*)Wv, (v4bf*)wvb, (int)(NW / 4));
  cvt_f32_to_bf16<<<(int)(NW / 4 / 256), 256, 0, stream>>>((const float4*)Wo, (v4bf*)wob, (int)(NW / 4));

  // 2) projections (head-split layouts); V also lands fp32 in d_out
  dim3 ggrid(64, 16), gblk(32, 8);
  gemm_bf16_wmma<<<ggrid, gblk, 0, stream>>>(xb, wqb, nullptr, qb, 1);
  gemm_bf16_wmma<<<ggrid, gblk, 0, stream>>>(xb, wkb, nullptr, kb, 1);
  gemm_bf16_wmma<<<ggrid, gblk, 0, stream>>>(xb, wvb, v_out,  vb, 1);

  // 3) RoPE on q (bf16) and k (bf16 + fp32 out)
  const int ropeN = B_ * NH_ * S_ * 64;
  rope_inplace<<<ropeN / 256, 256, 0, stream>>>(qb, nullptr, cosp, sinp);
  rope_inplace<<<ropeN / 256, 256, 0, stream>>>(kb, k_out,   cosp, sinp);

  // 4) causal flash attention
  attn_fwd_wmma<<<dim3(64, 16), gblk, 0, stream>>>(qb, kb, vb, ob);

  // 5) output projection y = out * Wo^T (fp32 row-major into d_out)
  gemm_bf16_wmma<<<ggrid, gblk, 0, stream>>>(ob, wob, y_out, nullptr, 0);
}
